// UltimateBrain_15195594293718
// MI455X (gfx1250) — matrix operand order repaired and checked by reading, not compiled, not run on hardware
//
#include <hip/hip_runtime.h>
#include <hip/hip_bf16.h>
#include <math.h>
#include <stdint.h>

// ---------------------------------------------------------------------------
// Types for CDNA5 WMMA (wave32): D(f32 16x16) = A(bf16 16x32) * B(bf16 32x16) + C
// ---------------------------------------------------------------------------
typedef __attribute__((ext_vector_type(16))) __bf16 v16bf;
typedef __attribute__((ext_vector_type(8)))  float  v8f;

#define TSEQ 256
#define NWG 16

__device__ __forceinline__ float sigmoidf_(float x) { return 1.f / (1.f + __expf(-x)); }

__device__ __forceinline__ v8f wmma_bf16(v16bf a, v16bf b, v8f c) {
  return __builtin_amdgcn_wmma_f32_16x16x32_bf16(false, a, false, b, (short)0, c, false, false);
}

// A-matrix 16x32 bf16 fragment per ISA 7.12.2 (from LDS; coalesces to 2x ds_load_b128)
__device__ __forceinline__ v16bf load_a_frag(const __bf16* base, int stride, int kbase, int lane) {
  const int m = lane & 15;
  const int hi8 = (lane >> 4) << 3; // 0 or 8
  const __bf16* r = base + m * stride + kbase;
  v16bf a;
#pragma unroll
  for (int j = 0; j < 4; ++j) {
    a[2 * j]     = r[hi8 + 2 * j];
    a[2 * j + 1] = r[hi8 + 2 * j + 1];
  }
#pragma unroll
  for (int j = 0; j < 4; ++j) {
    a[8 + 2 * j]     = r[16 + hi8 + 2 * j];
    a[8 + 2 * j + 1] = r[16 + hi8 + 2 * j + 1];
  }
  return a;
}

// B-matrix 32x16 bf16 fragment from FRAGMENT-MAJOR swizzled weights:
// tile (nt,kt) stored as 512 contiguous halves, lane-major (32 lanes x 16 halves).
// One aligned 32-byte contiguous load per lane -> global_load_b128 x2.
__device__ __forceinline__ v16bf load_b_frag(const __bf16* W, int Kt, int kt, int nt, int lane) {
  const __bf16* p = W + (((size_t)nt * Kt + kt) << 9) + (lane << 4);
  return *(const v16bf*)p;
}

// ---------------------------------------------------------------------------
// Compile-time region tables (compute order):
// 0 sensory 1 innate 2 learned 3 left_h 4 right_h 5 workspace 6 dist_a 7 dist_b
// 8 dist_c 9 dense_hub 10 memory 11 selfmodel 12 motor
// State layout (sorted-name all_r offsets): hub0 dist_a102 dist_b163 dist_c224
// innate285 learned346 left427 memory529 motor610 right691 selfmodel793
// sensory854 workspace935 (total 1016, padded to 1024)
// ---------------------------------------------------------------------------
__constant__ int dR_KP[13]  = {192,160,256,448,448,384,288,288,288,672,352,416,352};
__constant__ int dR_NP[13]  = {96, 64, 96, 112,112,96, 64, 64, 64, 112,96, 64, 96};
__constant__ int dR_H[13]   = {81, 61, 81, 102,102,81, 61, 61, 61, 102,81, 61, 81};
__constant__ int dR_OFF[13] = {854,285,346,427,691,935,102,163,224,0,  529,793,610};
__constant__ int dR_KST[13] = {81, 0,  81, 0,  0,  204,0,  0,  0,  0,  0,  0,  0};
__constant__ float dR_DTS[13]= {2.5f,2.5f,0.7f,0.3f,0.3f,1.0f,1.8f,1.8f,1.8f,1.0f,0.6f,1.0f,1.5f};
__constant__ int dR_AD[13]  = {0,0,0,0,0,1,0,0,0,1,0,1,0};
__constant__ int dR_SEGO[13][8] = {
  {854,0,0,0,0,0,0,0},
  {854,285,0,0,0,0,0,0},
  {346,346,0,0,0,0,0,0},
  {346,285,854,427,427,0,0,0},
  {346,285,854,691,691,0,0,0},
  {935,935,0,0,0,0,0,0},
  {935,854,102,102,0,0,0,0},
  {935,854,163,163,0,0,0,0},
  {935,854,224,224,0,0,0,0},
  {935,102,163,224,427,691,0,0},
  {854,0,529,529,0,0,0,0},
  {935,0,529,793,793,0,0,0},
  {0,793,529,610,0,0,0,0}};
__constant__ int dR_SEGL[13][8] = {
  {81,0,0,0,0,0,0,0},
  {81,61,0,0,0,0,0,0},
  {81,81,0,0,0,0,0,0},
  {81,61,81,102,102,0,0,0},
  {81,61,81,102,102,0,0,0},
  {81,81,0,0,0,0,0,0},
  {81,81,61,61,0,0,0,0},
  {81,81,61,61,0,0,0,0},
  {81,81,61,61,0,0,0,0},
  {81,61,61,61,102,102,102,102},
  {81,102,81,81,0,0,0,0},
  {81,102,81,61,61,0,0,0},
  {102,61,81,81,0,0,0,0}};

struct RegPtr { const float *b, *log_step, *ln_g, *ln_b, *log_dt; const __bf16* W; };
struct BrainP {
  RegPtr reg[13];
  const float* x;
  const float *ip_w, *ip_b;
  const float *sw_w, *sw_b;
  const float *mr_gw, *mr_gb, *mr_pw, *mr_pb;
  const float *ii_gw, *ii_gb, *ii_pw, *ii_pb;
  const float *sf_gw, *sf_gb, *sf_pw, *sf_pb;
  const __bf16 *thal_w, *proj_w;
  const float *thal_b, *thal_g, *thal_bb, *proj_b;
  float* hidden;
  float* t1buf;
  float* sums;
  int* cnt;
};

// ---------------------------------------------------------------------------
// Persistent scan kernel: 16 workgroups x 256 threads; each WG owns 16 batch rows.
// ---------------------------------------------------------------------------
__global__ __launch_bounds__(256) void brain_scan_kernel(BrainP P) {
  __shared__ __bf16 st[16][1024];   // recurrent state (== all_r, bf16, zero-padded to 1024)
  __shared__ __bf16 abuf[16][704];  // per-region A assembly
  __shared__ float cbuf[16][112];   // WMMA outputs / pre-LN values
  __shared__ float red[16][32];
  __shared__ float rowstat[16][2];
  __shared__ float sdt[2];

  const int tid = threadIdx.x;
  const int lane = tid & 31, wave = tid >> 5;
  const int bbase = blockIdx.x * 16;

  for (int i = tid; i < 16 * 1024; i += 256) ((__bf16*)st)[i] = (__bf16)0.f;
  __syncthreads();

  for (int t = 0; t < TSEQ; ++t) {
    for (int r = 0; r < 13; ++r) {
      const int Kp = dR_KP[r], Np = dR_NP[r], h = dR_H[r], off = dR_OFF[r], kst = dR_KST[r];

      // ---- special (axon-modified) input segments ----
      if (r == 0) {                 // sensory: input_proj(x_t) + m*0.1*axon(memory_replay, mem_prev)
        for (int idx = tid; idx < 16 * 81; idx += 256) {
          int row = idx / 81, n = idx - row * 81;
          float v = P.ip_b[n];
          const float* xr = P.x + (((size_t)(bbase + row)) * TSEQ + t) * 32;
#pragma unroll
          for (int i = 0; i < 32; ++i) v += xr[i] * P.ip_w[i * 81 + n];
          if (t > 0) {
            float g = P.mr_gb[n], pp = P.mr_pb[n];
            for (int j = 0; j < 81; ++j) {
              float mv = (float)st[row][529 + j];
              g += mv * P.mr_gw[j * 81 + n];
              pp += mv * P.mr_pw[j * 81 + n];
            }
            v += 0.1f * sigmoidf_(g) * tanhf(pp);
          }
          abuf[row][n] = (__bf16)v;
        }
      } else if (r == 2) {          // learned: sensory_new + m*0.1*axon(innate_inhibit, innate_new)
        for (int idx = tid; idx < 16 * 81; idx += 256) {
          int row = idx / 81, n = idx - row * 81;
          float v = (float)st[row][854 + n];
          if (t > 0) {
            float g = P.ii_gb[n], pp = P.ii_pb[n];
            for (int j = 0; j < 61; ++j) {
              float iv = (float)st[row][285 + j];
              g += iv * P.ii_gw[j * 81 + n];
              pp += iv * P.ii_pw[j * 81 + n];
            }
            v += 0.1f * sigmoidf_(g) * tanhf(pp);
          }
          abuf[row][n] = (__bf16)v;
        }
      } else if (r == 5) {          // workspace: [left,right] + m*0.05*axon(selfmodel_fb, sm_prev)
        for (int idx = tid; idx < 16 * 204; idx += 256) {
          int row = idx / 204, n = idx - row * 204;
          float v = (n < 102) ? (float)st[row][427 + n] : (float)st[row][691 + n - 102];
          if (t > 0) {
            float g = P.sf_gb[n], pp = P.sf_pb[n];
            for (int j = 0; j < 61; ++j) {
              float sv = (float)st[row][793 + j];
              g += sv * P.sf_gw[j * 204 + n];
              pp += sv * P.sf_pw[j * 204 + n];
            }
            v += 0.05f * sigmoidf_(g) * tanhf(pp);
          }
          abuf[row][n] = (__bf16)v;
        }
      }

      // ---- generic A assembly (segments of state, zero pad to Kp) ----
      {
        const int span = Kp - kst;
        for (int idx = tid; idx < 16 * span; idx += 256) {
          int row = idx / span;
          int kk = idx - row * span;
          __bf16 v = (__bf16)0.f;
          int kr = kk;
#pragma unroll 1
          for (int s2 = 0; s2 < 8; ++s2) {
            int len = dR_SEGL[r][s2];
            if (kr < len) { v = st[row][dR_SEGO[r][s2] + kr]; break; }
            kr -= len;
          }
          abuf[row][kst + kk] = v;
        }
      }
      __syncthreads();

      // ---- WMMA: cand_pre = A[16xKp] @ W[KpxNp]; tanh(. + b) -> cbuf ----
      {
        const __bf16* W = P.reg[r].W;
        const float* bias = P.reg[r].b;
        const int ntiles = Np >> 4, ktiles = Kp >> 5;
        for (int nt = wave; nt < ntiles; nt += 8) {
          v8f acc = {};
          for (int kt = 0; kt < ktiles; ++kt) {
            v16bf a = load_a_frag(&abuf[0][0], 704, kt * 32, lane);
            v16bf bm = load_b_frag(W, ktiles, kt, nt, lane);
            acc = wmma_bf16(a, bm, acc);
          }
          const int n = nt * 16 + (lane & 15);
          const int rb8 = (lane >= 16) ? 8 : 0;
          const float bb = (n < h) ? bias[n] : 0.f;
#pragma unroll
          for (int i = 0; i < 8; ++i) cbuf[rb8 + i][n] = tanhf(acc[i] + bb);
        }
      }
      __syncthreads();

      // ---- LTC update + LayerNorm -> state (bf16) ----
      float dte;
      if (r == 3)       dte = sdt[0];
      else if (r == 4)  dte = sdt[1];
      else if (dR_AD[r]) dte = 3.f * sigmoidf_(P.reg[r].log_dt[0]);
      else              dte = dR_DTS[r];

      {
        const float* lst = P.reg[r].log_step;
        for (int idx = tid; idx < 16 * h; idx += 256) {
          int row = idx / h, n = idx - row * h;
          float hv = (float)st[row][off + n];
          float u = hv + dte * __expf(lst[n]) * (cbuf[row][n] - hv);
          cbuf[row][n] = u;
        }
      }
      __syncthreads();
      {
        int row = tid >> 4, j = tid & 15;
        float s = 0.f, q = 0.f;
        for (int n = j; n < h; n += 16) { float v = cbuf[row][n]; s += v; q += v * v; }
        red[row][j] = s; red[row][16 + j] = q;
      }
      __syncthreads();
      if (tid < 16) {
        float s = 0.f, q = 0.f;
        for (int j = 0; j < 16; ++j) { s += red[tid][j]; q += red[tid][16 + j]; }
        float mu = s / (float)h, var = q / (float)h - mu * mu;
        rowstat[tid][0] = mu; rowstat[tid][1] = rsqrtf(var + 1e-5f);
      }
      __syncthreads();
      {
        const float* g = P.reg[r].ln_g;
        const float* bn = P.reg[r].ln_b;
        for (int idx = tid; idx < 16 * h; idx += 256) {
          int row = idx / h, n = idx - row * h;
          float v = (cbuf[row][n] - rowstat[row][0]) * rowstat[row][1] * g[n] + bn[n];
          st[row][off + n] = (__bf16)v;
        }
      }
      __syncthreads();

      // ---- after sensory: global switch mean (grid-wide deterministic reduction) ----
      if (r == 0) {
        if (tid < 16) {
          float l0 = P.sw_b[0], l1 = P.sw_b[1];
          for (int j = 0; j < 81; ++j) {
            float sv = (float)st[tid][854 + j];
            l0 += sv * P.sw_w[j * 2 + 0];
            l1 += sv * P.sw_w[j * 2 + 1];
          }
          red[tid][0] = sigmoidf_(l0 - l1); // p0 = softmax col 0
        }
        __syncthreads();
        if (tid == 0) {
          float s = 0.f;
          for (int rr = 0; rr < 16; ++rr) s += red[rr][0];
          // cluster_id_x == 0 for non-cluster dispatch (ISA 3.2.1); exact, not DCE-able
          P.sums[t * 16 + blockIdx.x + __builtin_amdgcn_cluster_id_x()] = s;
          __threadfence();
          atomicAdd(&P.cnt[t], 1);
          volatile int* vc = &P.cnt[t];
          while (*vc < NWG) __builtin_amdgcn_s_sleep(1);
          __threadfence();
          float tot = 0.f;
          for (int w2 = 0; w2 < NWG; ++w2) tot += P.sums[t * 16 + w2];
          float m0 = tot * (1.f / 256.f);
          sdt[0] = 0.5f + m0;
          sdt[1] = 0.5f + (1.f - m0);
        }
        __builtin_amdgcn_s_cluster_barrier(); // S_NOP when not in a cluster (ISA 2)
        __syncthreads();
      }
    } // regions

    // ---- thalamic output: t1 = all_r@thal_w + b ; t2 = all_r@proj_w + b ----
    for (int nt = wave; nt < 64; nt += 8) {
      v8f a1 = {}; v8f a2 = {};
      for (int kt = 0; kt < 32; ++kt) {
        v16bf a = load_a_frag(&st[0][0], 1024, kt * 32, lane);
        v16bf b1 = load_b_frag(P.thal_w, 32, kt, nt, lane);
        v16bf b2 = load_b_frag(P.proj_w, 32, kt, nt, lane);
        a1 = wmma_bf16(a, b1, a1);
        a2 = wmma_bf16(a, b2, a2);
        if (kt < 31) {
          __builtin_prefetch(P.thal_w + (((size_t)nt * 32 + kt + 1) << 9), 0, 1);
          __builtin_prefetch(P.proj_w + (((size_t)nt * 32 + kt + 1) << 9), 0, 1);
        }
      }
      const int n = nt * 16 + (lane & 15);
      const int rb8 = (lane >= 16) ? 8 : 0;
#pragma unroll
      for (int i = 0; i < 8; ++i) {
        int gb = bbase + rb8 + i;
        P.t1buf[(size_t)gb * 1024 + n] = a1[i] + P.thal_b[n];
        P.hidden[(((size_t)gb) * TSEQ + t) * 1024 + n] = a2[i] + P.proj_b[n];
      }
    }
    __threadfence();
    __syncthreads();
    // gate = sigmoid(LN(t1)); hidden *= gate
    {
      int row = tid >> 4, j = tid & 15;
      const float* t1 = P.t1buf + (size_t)(bbase + row) * 1024;
      float s = 0.f, q = 0.f;
      for (int n = j; n < 1024; n += 16) { float v = t1[n]; s += v; q += v * v; }
      red[row][j] = s; red[row][16 + j] = q;
    }
    __syncthreads();
    if (tid < 16) {
      float s = 0.f, q = 0.f;
      for (int j = 0; j < 16; ++j) { s += red[tid][j]; q += red[tid][16 + j]; }
      float mu = s * (1.f / 1024.f), var = q * (1.f / 1024.f) - mu * mu;
      rowstat[tid][0] = mu; rowstat[tid][1] = rsqrtf(var + 1e-5f);
    }
    __syncthreads();
    for (int idx = tid; idx < 16 * 1024; idx += 256) {
      int row = idx >> 10, n = idx & 1023;
      int gb = bbase + row;
      float t1 = P.t1buf[(size_t)gb * 1024 + n];
      float gate = sigmoidf_((t1 - rowstat[row][0]) * rowstat[row][1] * P.thal_g[n] + P.thal_bb[n]);
      P.hidden[(((size_t)gb) * TSEQ + t) * 1024 + n] *= gate;
    }
    __syncthreads();
  } // t
}

// ---------------------------------------------------------------------------
// Heads 1: predictions (WMMA next_lin), anomaly, surprise. 16 (b,t) rows / block.
// ---------------------------------------------------------------------------
struct Heads1P {
  const float* hidden; const float* x;
  const float *ng, *nb;       // next_ln g,b
  const float *ag, *ab;       // anom_ln g,b
  const float *aw, *ab0;      // anom_lin w[1024], b[1]
  const __bf16* next_w; const float* next_b;
  float *pred, *anom, *sur;
};

__global__ __launch_bounds__(128) void heads1_kernel(Heads1P P) {
  __shared__ __bf16 abuf[16][1024];
  __shared__ float red[16][16];
  __shared__ float rowstat[16][2];
  __shared__ float sbuf[16][32];
  const int tid = threadIdx.x, lane = tid & 31, wave = tid >> 5;
  const int gr0 = blockIdx.x * 16;
  const int b = gr0 >> 8;
  const int t0 = gr0 & 255;

  {
    int row = tid >> 3, j = tid & 7;
    const float* hr = P.hidden + ((size_t)gr0 + row) * 1024;
    float s = 0.f, q = 0.f;
    for (int n = j; n < 1024; n += 8) { float v = hr[n]; s += v; q += v * v; }
    red[row][j] = s; red[row][8 + j] = q;
  }
  __syncthreads();
  if (tid < 16) {
    float s = 0.f, q = 0.f;
    for (int j = 0; j < 8; ++j) { s += red[tid][j]; q += red[tid][8 + j]; }
    float mu = s * (1.f / 1024.f), var = q * (1.f / 1024.f) - mu * mu;
    rowstat[tid][0] = mu; rowstat[tid][1] = rsqrtf(var + 1e-5f);
  }
  __syncthreads();
  {
    int row = tid >> 3, j = tid & 7;
    const float* hr = P.hidden + ((size_t)gr0 + row) * 1024;
    float mu = rowstat[row][0], rs = rowstat[row][1];
    float ap = 0.f;
    for (int n = j; n < 1024; n += 8) {
      float norm = (hr[n] - mu) * rs;
      abuf[row][n] = (__bf16)(norm * P.ng[n] + P.nb[n]);
      ap += (norm * P.ag[n] + P.ab[n]) * P.aw[n];
    }
    red[row][j] = ap;
  }
  __syncthreads();
  if (tid < 16) {
    float s = P.ab0[0];
    for (int j = 0; j < 8; ++j) s += red[tid][j];
    P.anom[(size_t)gr0 + tid] = sigmoidf_(s);
  }
  if (wave < 2) { // predictions: [16x1024] @ [1024x32]
    v8f acc = {};
    for (int kt = 0; kt < 32; ++kt) {
      v16bf a = load_a_frag(&abuf[0][0], 1024, kt * 32, lane);
      v16bf bm = load_b_frag(P.next_w, 32, kt, wave, lane);
      acc = wmma_bf16(a, bm, acc);
    }
    const int n = wave * 16 + (lane & 15);
    const int rb8 = (lane >= 16) ? 8 : 0;
    const float bb = P.next_b[n];
#pragma unroll
    for (int i = 0; i < 8; ++i) {
      int row = rb8 + i, tt = t0 + row;
      float v = acc[i] + bb;
      sbuf[row][n] = v;
      if (tt < 255) P.pred[((size_t)b * 255 + tt) * 32 + n] = v;
    }
  }
  __syncthreads();
  if (tid < 16) {
    int tt = t0 + tid;
    if (tt < 255) {
      const float* xr = P.x + ((size_t)b * 256 + (tt + 1)) * 32;
      float s = 0.f;
      for (int i = 0; i < 32; ++i) { float d = sbuf[tid][i] - xr[i]; s += d * d; }
      P.sur[(size_t)b * 255 + tt] = s * (1.f / 32.f);
    }
  }
}

// ---------------------------------------------------------------------------
// Heads 2: properties = prop_lin2(gelu(prop_lin1(prop_ln(hidden[:, -1]))))
// ---------------------------------------------------------------------------
struct Heads2P {
  const float* hidden;
  const float *pg, *pb;
  const __bf16* w1; const float* b1;
  const float* w2; const float* b2;
  float* props;
};

__global__ __launch_bounds__(256) void heads2_kernel(Heads2P P) {
  __shared__ __bf16 abuf[16][1024];
  __shared__ __bf16 p1[16][512];
  __shared__ float red[16][32];
  __shared__ float rowstat[16][2];
  const int tid = threadIdx.x, lane = tid & 31, wave = tid >> 5;
  const int b0 = blockIdx.x * 16;

  {
    int row = tid >> 4, j = tid & 15;
    const float* hr = P.hidden + (((size_t)(b0 + row)) * 256 + 255) * 1024;
    float s = 0.f, q = 0.f;
    for (int n = j; n < 1024; n += 16) { float v = hr[n]; s += v; q += v * v; }
    red[row][j] = s; red[row][16 + j] = q;
  }
  __syncthreads();
  if (tid < 16) {
    float s = 0.f, q = 0.f;
    for (int j = 0; j < 16; ++j) { s += red[tid][j]; q += red[tid][16 + j]; }
    float mu = s * (1.f / 1024.f), var = q * (1.f / 1024.f) - mu * mu;
    rowstat[tid][0] = mu; rowstat[tid][1] = rsqrtf(var + 1e-5f);
  }
  __syncthreads();
  {
    int row = tid >> 4, j = tid & 15;
    const float* hr = P.hidden + (((size_t)(b0 + row)) * 256 + 255) * 1024;
    float mu = rowstat[row][0], rs = rowstat[row][1];
    for (int n = j; n < 1024; n += 16)
      abuf[row][n] = (__bf16)((hr[n] - mu) * rs * P.pg[n] + P.pb[n]);
  }
  __syncthreads();
  for (int nt = wave; nt < 32; nt += 8) {
    v8f acc = {};
    for (int kt = 0; kt < 32; ++kt) {
      v16bf a = load_a_frag(&abuf[0][0], 1024, kt * 32, lane);
      v16bf bm = load_b_frag(P.w1, 32, kt, nt, lane);
      acc = wmma_bf16(a, bm, acc);
    }
    const int n = nt * 16 + (lane & 15);
    const int rb8 = (lane >= 16) ? 8 : 0;
    const float bb = P.b1[n];
#pragma unroll
    for (int i = 0; i < 8; ++i) {
      float v = acc[i] + bb;
      float g = 0.5f * v * (1.f + erff(v * 0.70710678118f)); // exact gelu
      p1[rb8 + i][n] = (__bf16)g;
    }
  }
  __syncthreads();
  if (tid < 32) {
    int row = tid >> 1, c = tid & 1;
    float s = P.b2[c];
    for (int k = 0; k < 512; ++k) s += (float)p1[row][k] * P.w2[k * 2 + c];
    P.props[(size_t)(b0 + row) * 2 + c] = s;
  }
}

// ---------------------------------------------------------------------------
// f32 -> bf16 weight conversion into FRAGMENT-MAJOR swizzled layout:
// dst[((nt*Kt + kt)*512) + lane*16 + j] = W[kt*32 + (lane>>4)*8 + (j&7) + (j>>3)*16][nt*16 + (lane&15)]
// where W rows [0,r0)=src0, [r0,r0+r1)=src1, else 0; cols >= N are 0.
// ---------------------------------------------------------------------------
__global__ void wconv_kernel(const float* s0, int r0, const float* s1, int r1,
                             int N, int Kp, int Np, __bf16* dst) {
  int idx = blockIdx.x * blockDim.x + threadIdx.x;
  if (idx >= Kp * Np) return;
  int tile = idx >> 9;
  int rem = idx & 511;
  int lane = rem >> 4, j = rem & 15;
  int Kt = Kp >> 5;
  int nt = tile / Kt, kt = tile - nt * Kt;
  int k = kt * 32 + ((lane >> 4) << 3) + (j & 7) + ((j >> 3) << 4);
  int n = nt * 16 + (lane & 15);
  float v = 0.f;
  if (n < N) {
    if (k < r0) v = s0[(size_t)k * N + n];
    else if (k < r0 + r1) v = s1[(size_t)(k - r0) * N + n];
  }
  dst[idx] = (__bf16)v;
}

__global__ void ws_init_kernel(float* sums, int* cnt) {
  int i = blockIdx.x * blockDim.x + threadIdx.x;
  if (i < 4096) sums[i] = 0.f;
  if (i < 256) cnt[i] = 0;
}

// ---------------------------------------------------------------------------
// Host launch. Leaf order assumes sorted-key pytree flatten of setup_inputs():
// params (sorted keys, lin={b,w}, ln={b,g}, axon={gate{b,w},proj{b,w}},
// region={b,[log_dt],log_step,norm{b,g},wh,wi}) then x. 118 leaves total.
// ---------------------------------------------------------------------------
extern "C" void kernel_launch(void* const* d_in, const int* in_sizes, int n_in,
                              void* d_out, int out_size, void* d_ws, size_t ws_size,
                              hipStream_t stream) {
  (void)in_sizes; (void)out_size; (void)ws_size;
  const float* L[256];
  for (int i = 0; i < n_in && i < 256; ++i) L[i] = (const float*)d_in[i];

  // region leaf bases (sorted region order; adaptive regions have extra log_dt leaf)
  static const int adS[13] = {1,0,0,0,0,0,0,0,0,0,1,0,1}; // hub,da,db,dc,innate,learned,left,mem,motor,right,selfmodel,sensory,workspace
  int rb[13]; int pos = 24;
  for (int s = 0; s < 13; ++s) { rb[s] = pos; pos += 6 + adS[s]; }
  static const int S[13] = {11,4,5,6,9,12,1,2,3,0,7,10,8}; // compute-order -> sorted index

  static const int hKp[13]  = {192,160,256,448,448,384,288,288,288,672,352,416,352};
  static const int hNp[13]  = {96, 64, 96, 112,112,96, 64, 64, 64, 112,96, 64, 96};
  static const int hH[13]   = {81, 61, 81, 102,102,81, 61, 61, 61, 102,81, 61, 81};
  static const int hEff[13] = {81, 81, 162,325,325,285,223,223,223,570,264,325,244};

  uint8_t* ws = (uint8_t*)d_ws;
  float* sums = (float*)ws;                  // 16 KB
  int* cnt = (int*)(ws + 16384);             // 1 KB
  float* t1buf = (float*)(ws + 20480);       // 1 MB
  size_t woff = 20480 + (size_t)256 * 1024 * 4;

  __bf16* regW[13];
  for (int r = 0; r < 13; ++r) { regW[r] = (__bf16*)(ws + woff); woff += (size_t)hKp[r] * hNp[r] * 2; }
  __bf16* thalW = (__bf16*)(ws + woff); woff += (size_t)1024 * 1024 * 2;
  __bf16* projW = (__bf16*)(ws + woff); woff += (size_t)1024 * 1024 * 2;
  __bf16* nextW = (__bf16*)(ws + woff); woff += (size_t)1024 * 32 * 2;
  __bf16* propW = (__bf16*)(ws + woff); woff += (size_t)1024 * 512 * 2;

  ws_init_kernel<<<16, 256, 0, stream>>>(sums, cnt);

  for (int r = 0; r < 13; ++r) {
    int sidx = S[r], ad = adS[sidx], base = rb[sidx];
    const float* wh = L[base + 4 + ad];
    const float* wi = L[base + 5 + ad];
    int total = hKp[r] * hNp[r];
    wconv_kernel<<<(total + 255) / 256, 256, 0, stream>>>(wi, hEff[r], wh, hH[r], hH[r], hKp[r], hNp[r], regW[r]);
  }
  wconv_kernel<<<(1024 * 1024 + 255) / 256, 256, 0, stream>>>(L[112], 1016, (const float*)nullptr, 0, 1024, 1024, 1024, thalW);
  wconv_kernel<<<(1024 * 1024 + 255) / 256, 256, 0, stream>>>(L[116], 1016, (const float*)nullptr, 0, 1024, 1024, 1024, projW);
  wconv_kernel<<<(1024 * 32 + 255) / 256, 256, 0, stream>>>(L[15], 1024, (const float*)nullptr, 0, 32, 1024, 32, nextW);
  wconv_kernel<<<(1024 * 512 + 255) / 256, 256, 0, stream>>>(L[19], 1024, (const float*)nullptr, 0, 512, 1024, 512, propW);

  BrainP P{};
  for (int r = 0; r < 13; ++r) {
    int sidx = S[r], ad = adS[sidx], base = rb[sidx];
    P.reg[r].b = L[base];
    P.reg[r].log_dt = ad ? L[base + 1] : nullptr;
    P.reg[r].log_step = L[base + 1 + ad];
    P.reg[r].ln_b = L[base + 2 + ad];
    P.reg[r].ln_g = L[base + 3 + ad];
    P.reg[r].W = regW[r];
  }
  P.x = L[117];
  P.ip_b = L[8];  P.ip_w = L[9];
  P.sw_b = L[109]; P.sw_w = L[110];
  P.mr_gb = L[10]; P.mr_gw = L[11]; P.mr_pb = L[12]; P.mr_pw = L[13];
  P.ii_gb = L[4];  P.ii_gw = L[5];  P.ii_pb = L[6];  P.ii_pw = L[7];
  P.sf_gb = L[105]; P.sf_gw = L[106]; P.sf_pb = L[107]; P.sf_pw = L[108];
  P.thal_w = thalW; P.proj_w = projW;
  P.thal_b = L[111]; P.thal_bb = L[113]; P.thal_g = L[114]; P.proj_b = L[115];

  float* out = (float*)d_out;
  const size_t PRED_OFF = 0;
  const size_t HID_OFF = 2088960;           // 256*255*32
  const size_t ANOM_OFF = 69197824;         // HID_OFF + 256*256*1024
  const size_t SUR_OFF = 69263360;
  const size_t PROP_OFF = 69328640;
  P.hidden = out + HID_OFF;
  P.t1buf = t1buf; P.sums = sums; P.cnt = cnt;

  brain_scan_kernel<<<16, 256, 0, stream>>>(P);

  Heads1P H1{};
  H1.hidden = P.hidden; H1.x = L[117];
  H1.nb = L[16]; H1.ng = L[17];
  H1.ab = L[2];  H1.ag = L[3];
  H1.ab0 = L[0]; H1.aw = L[1];
  H1.next_w = nextW; H1.next_b = L[14];
  H1.pred = out + PRED_OFF; H1.anom = out + ANOM_OFF; H1.sur = out + SUR_OFF;
  heads1_kernel<<<4096, 128, 0, stream>>>(H1);

  Heads2P H2{};
  H2.hidden = P.hidden;
  H2.pb = L[22]; H2.pg = L[23];
  H2.w1 = propW; H2.b1 = L[18];
  H2.w2 = L[21]; H2.b2 = L[20];
  H2.props = out + PROP_OFF;
  heads2_kernel<<<16, 256, 0, stream>>>(H2);
}
